// POS_LSTM_CHAR_31250182045957
// MI455X (gfx1250) — matrix operand order repaired and checked by reading, not compiled, not run on hardware
//
#include <hip/hip_runtime.h>
#include <hip/hip_bf16.h>

// ---------------------------------------------------------------------------
// POS tagger: char-LSTM -> word-LSTM -> linear.   S=1024, T=16, DC=64, DW=128,
// H=512, TGT=50.  Dense GEMMs on v_wmma_f32_16x16x32_bf16; recurrent weights
// staged into LDS with the Tensor Data Mover; per-step hidden-state broadcast
// via global_load_async_to_lds (ASYNCcnt path).
// ---------------------------------------------------------------------------

typedef __attribute__((ext_vector_type(16))) __bf16 v16bf;
typedef __attribute__((ext_vector_type(8)))  float  v8f;
typedef __attribute__((ext_vector_type(4)))  unsigned v4u;
typedef __attribute__((ext_vector_type(8)))  int      v8i;
typedef __attribute__((ext_vector_type(4)))  int      v4i;

#define S_WORDS 1024
#define T_CHARS 16
#define DC 64
#define DW 128
#define HW 512
#define TGT 50

// ----- workspace layout (bytes) -----
#define OFF_BC   ((size_t)0)          // bf16 [128][256]   char weights^T (x|h)
#define OFF_BWX  ((size_t)65536)      // bf16 [192][2048]  Ww_ih^T
#define OFF_BO   ((size_t)851968)     // bf16 [512][64]    W_out^T (padded 50->64)
#define OFF_WIN  ((size_t)917504)     // bf16 [1024][192]  word-lstm inputs (w|char_feat)
#define OFF_GX   ((size_t)1310720)    // f32  [1024][2048] precomputed x-gates + bw
#define OFF_HS   ((size_t)9699328)    // bf16 [1024][512]  word-lstm hidden states
#define OFF_CTR  ((size_t)10747904)   // u32  grid-sync counter
#define OFF_WHH  ((size_t)10748160)   // bf16 [2048][512]  Ww_hh (TDM source)

__device__ __forceinline__ float sigmoidf(float x) {
    return 1.0f / (1.0f + __expf(-x));
}

// --- WMMA fragment loaders (gfx1250 wave32 16-bit layouts, ISA 7.12.2) ------
__device__ __forceinline__ v16bf loadA_rm(const __bf16* src, int ld, int k0) {
    int lane = threadIdx.x & 31;
    int m = lane & 15, half = lane >> 4;
    v16bf a;
#pragma unroll
    for (int e = 0; e < 16; ++e) {
        int p = e >> 1, w = e & 1;
        int k = ((p & 3) * 2 + w) + 8 * half + 16 * (p >> 2);
        a[e] = src[m * ld + k0 + k];
    }
    return a;
}
__device__ __forceinline__ v16bf loadB_rm(const __bf16* src, int ld, int k0, int n0) {
    int lane = threadIdx.x & 31;
    int n = lane & 15, half = lane >> 4;
    v16bf b;
#pragma unroll
    for (int e = 0; e < 16; ++e) {
        int p = e >> 1, w = e & 1;
        int k = ((p & 3) * 2 + w) + 8 * half + 16 * (p >> 2);
        b[e] = src[(k0 + k) * ld + n0 + n];
    }
    return b;
}

// --- Tensor Data Mover: 2D bf16 tile (tile_d1 rows x tile_d0 cols) ----------
// Descriptor bit packing per CDNA5 ISA §8.3 (group 0) / §8.4 (group 1).
__device__ __forceinline__ void tdm_load_tile_2d(
    unsigned lds_byte_addr, const __bf16* tile_start,
    unsigned tensor_d0, unsigned tensor_d1, unsigned stride0,
    unsigned tile_d0, unsigned tile_d1)
{
    unsigned long long ga = (unsigned long long)(size_t)tile_start;
    v4u g0;
    g0[0] = 1u;                                             // count=1, user mode
    g0[1] = lds_byte_addr;                                  // lds_addr
    g0[2] = (unsigned)(ga & 0xFFFFFFFFu);                   // global_addr[31:0]
    g0[3] = (unsigned)((ga >> 32) & 0x01FFFFFFu)            // global_addr[56:32]
          | (2u << 30);                                     // type=2 ("image")
    v8i g1;
    g1[0] = (int)(1u << 16);                                // data_size=1 (2 bytes)
    g1[1] = (int)((tensor_d0 & 0xFFFFu) << 16);             // tensor_dim0[15:0] @bit48
    g1[2] = (int)(((tensor_d0 >> 16) & 0xFFFFu)             // tensor_dim0[31:16]
          | ((tensor_d1 & 0xFFFFu) << 16));                 // tensor_dim1[15:0]
    g1[3] = (int)(((tensor_d1 >> 16) & 0xFFFFu)             // tensor_dim1[31:16]
          | ((tile_d0 & 0xFFFFu) << 16));                   // tile_dim0
    g1[4] = (int)(tile_d1 & 0xFFFFu);                       // tile_dim1 (tile_dim2=0)
    g1[5] = (int)stride0;                                   // tensor_dim0_stride[31:0]
    g1[6] = 0;                                              // stride0[47:32], stride1 lo
    g1[7] = 0;
    v4i z4 = {};
#if defined(__clang_major__) && (__clang_major__ >= 23)
    v8i z8 = {};
    __builtin_amdgcn_tensor_load_to_lds(g0, g1, z4, z4, z8, 0);
#else
    __builtin_amdgcn_tensor_load_to_lds(g0, g1, z4, z4, 0);
#endif
}

// ---------------------------------------------------------------------------
// Prep kernels
// ---------------------------------------------------------------------------
__global__ __launch_bounds__(256) void prep_bc(const float* Wc_ih, const float* Wc_hh,
                                               __bf16* Bc) {
    int i = blockIdx.x * 256 + threadIdx.x;          // 128*256
    if (i >= 128 * 256) return;
    int k = i >> 8, n = i & 255;
    float v = (k < DC) ? Wc_ih[n * DC + k] : Wc_hh[n * DC + (k - DC)];
    Bc[k * 256 + n] = (__bf16)v;
}

__global__ __launch_bounds__(256) void prep_bwx(const float* Ww_ih, __bf16* Bwx) {
    int i = blockIdx.x * 256 + threadIdx.x;          // 192*2048
    if (i >= 192 * 2048) return;
    int k = i >> 11, n = i & 2047;
    Bwx[k * 2048 + n] = (__bf16)Ww_ih[n * 192 + k];
}

__global__ __launch_bounds__(256) void prep_bo(const float* W_out, __bf16* Bo) {
    int i = blockIdx.x * 256 + threadIdx.x;          // 512*64
    if (i >= 512 * 64) return;
    int k = i >> 6, n = i & 63;
    Bo[k * 64 + n] = (__bf16)((n < TGT) ? W_out[n * HW + k] : 0.0f);
}

__global__ __launch_bounds__(256) void prep_win(const int* word_ids, const float* word_emb,
                                                __bf16* WIN) {
    int i = blockIdx.x * 256 + threadIdx.x;          // 1024*128
    if (i >= S_WORDS * DW) return;
    int s = i >> 7, k = i & 127;
    WIN[s * 192 + k] = (__bf16)word_emb[(size_t)word_ids[s] * DW + k];
}

__global__ __launch_bounds__(256) void prep_whh(const float* Ww_hh, __bf16* Whh_bf) {
    int i = blockIdx.x * 256 + threadIdx.x;          // 2048*512
    if (i >= 2048 * HW) return;
    Whh_bf[i] = (__bf16)Ww_hh[i];
}

__global__ void prep_ctr(unsigned* ctr) { if (threadIdx.x == 0) *ctr = 0u; }

// ---------------------------------------------------------------------------
// Kernel 1: char LSTM. One WG = 16 words, 8 waves; B fragments resident in
// VGPRs across all 16 timesteps; last-valid h fused into WIN[:,128:192].
// ---------------------------------------------------------------------------
__global__ __launch_bounds__(256) void char_lstm_kernel(
    const int* __restrict__ char_ids, const int* __restrict__ char_lens,
    const float* __restrict__ char_emb, const __bf16* __restrict__ Bc,
    const float* __restrict__ bc, __bf16* __restrict__ WIN) {
    __shared__ __bf16 Abuf[16][128];      // [word][x(64)|h(64)]
    __shared__ float  gates[16][256];
    __shared__ float  cbuf[16][64];

    int tid = threadIdx.x;
    int wave = tid >> 5, lane = tid & 31;
    int s0 = blockIdx.x * 16;

    for (int i = tid; i < 16 * 64; i += 256) {
        int m = i >> 6, j = i & 63;
        Abuf[m][64 + j] = (__bf16)0.0f;
        cbuf[m][j] = 0.0f;
    }
    v16bf breg[2][4];
#pragma unroll
    for (int tt = 0; tt < 2; ++tt) {
        int nt = wave * 2 + tt;
#pragma unroll
        for (int ks = 0; ks < 4; ++ks)
            breg[tt][ks] = loadB_rm(Bc, 256, ks * 32, nt * 16);
    }
    __syncthreads();

    for (int t = 0; t < T_CHARS; ++t) {
        for (int i = tid; i < 16 * 64; i += 256) {           // gather x_t
            int m = i >> 6, k = i & 63;
            int cid = char_ids[(s0 + m) * T_CHARS + t];
            Abuf[m][k] = (__bf16)char_emb[cid * DC + k];
        }
        __syncthreads();
#pragma unroll
        for (int tt = 0; tt < 2; ++tt) {                     // gates GEMM
            int nt = wave * 2 + tt;
            v8f acc = {};
#pragma unroll
            for (int ks = 0; ks < 4; ++ks) {
                v16bf a = loadA_rm(&Abuf[0][0], 128, ks * 32);
                acc = __builtin_amdgcn_wmma_f32_16x16x32_bf16(
                    false, a, false, breg[tt][ks], (short)0, acc, false, false);
            }
            int n = lane & 15, half = lane >> 4;
#pragma unroll
            for (int r = 0; r < 8; ++r) {
                int m = r + 8 * half, col = nt * 16 + n;
                gates[m][col] = acc[r] + bc[col];
            }
        }
        __syncthreads();
        for (int i = tid; i < 16 * 64; i += 256) {           // cell update
            int m = i >> 6, j = i & 63;
            float ig = sigmoidf(gates[m][j]);
            float fg = sigmoidf(gates[m][64 + j]);
            float gg = tanhf(gates[m][128 + j]);
            float og = sigmoidf(gates[m][192 + j]);
            float c  = fg * cbuf[m][j] + ig * gg;
            float h  = og * tanhf(c);
            cbuf[m][j] = c;
            Abuf[m][64 + j] = (__bf16)h;
            if (t == char_lens[s0 + m] - 1)
                WIN[(s0 + m) * 192 + DW + j] = (__bf16)h;    // fused ragged gather
        }
        __syncthreads();
    }
}

// ---------------------------------------------------------------------------
// Kernel 2: Gx = WIN[1024x192] @ Ww_ih^T + bw   (pure WMMA GEMM, N=2048)
// ---------------------------------------------------------------------------
__global__ __launch_bounds__(256) void gx_kernel(
    const __bf16* __restrict__ WIN, const __bf16* __restrict__ Bwx,
    const float* __restrict__ bw, float* __restrict__ Gx) {
    int wid = blockIdx.x * 8 + (threadIdx.x >> 5);
    int lane = threadIdx.x & 31;
    int mt = wid >> 7, nt = wid & 127;    // 64 x 128 tiles
    v8f acc = {};
#pragma unroll
    for (int ks = 0; ks < 6; ++ks) {      // K = 192
        v16bf a = loadA_rm(WIN + (size_t)(mt * 16) * 192, 192, ks * 32);
        v16bf b = loadB_rm(Bwx, 2048, ks * 32, nt * 16);
        acc = __builtin_amdgcn_wmma_f32_16x16x32_bf16(
            false, a, false, b, (short)0, acc, false, false);
    }
    int n = lane & 15, half = lane >> 4;
#pragma unroll
    for (int r = 0; r < 8; ++r) {
        int m = mt * 16 + r + 8 * half, col = nt * 16 + n;
        Gx[(size_t)m * 2048 + col] = acc[r] + bw[col];
    }
}

// ---------------------------------------------------------------------------
// Kernel 3: word LSTM recurrence.  64 persistent cooperative WGs; recurrent
// weight slice (32x512 bf16) DMA'd into LDS by the Tensor Data Mover (4 tiles,
// one per gate block); per-step h broadcast via global_load_async_to_lds.
// ---------------------------------------------------------------------------
#define NWG 64
__global__ __launch_bounds__(256) void word_lstm_kernel(
    const __bf16* __restrict__ Whh_bf, const float* __restrict__ Gx,
    __bf16* __restrict__ hs_bf, unsigned* __restrict__ ctr) {
    __shared__ __bf16 Wl[32][512];        // 32 KB resident recurrent weights
    __shared__ __bf16 hbuf[512];
    __shared__ float  partial[32][8];
    __shared__ float  gv[32];
    __shared__ float  cst[8];

    int g = blockIdx.x, tid = threadIdx.x;

    // --- stage the 4 gate blocks (8 rows x 512 cols each) with the TDM ---
    if (tid < 32) {                        // wave 0 issues; EXEC ignored by TDM
#pragma unroll
        for (int gate = 0; gate < 4; ++gate) {
            unsigned lds_dst = (unsigned)(size_t)&Wl[gate * 8][0];
            const __bf16* src = Whh_bf + (size_t)(gate * HW + g * 8) * HW;
            tdm_load_tile_2d(lds_dst, src,
                             /*tensor_d0=*/HW, /*tensor_d1=*/2048,
                             /*stride0=*/HW, /*tile_d0=*/HW, /*tile_d1=*/8);
        }
        __builtin_amdgcn_s_wait_tensorcnt(0);
    }
    for (int i = tid; i < 512; i += 256) hbuf[i] = (__bf16)0.0f;
    if (tid < 8) cst[tid] = 0.0f;
    __syncthreads();

    for (int s = 0; s < S_WORDS; ++s) {
        if (s > 0) {
            // async global->LDS broadcast of h_{s-1} (1 KB), 4 B per lane
            unsigned ldst = (unsigned)(size_t)&hbuf[tid * 2];
            unsigned long long gsrc =
                (unsigned long long)(size_t)(hs_bf + (size_t)(s - 1) * HW + tid * 2);
            asm volatile("global_load_async_to_lds_b32 %0, %1, off"
                         :: "v"(ldst), "v"(gsrc) : "memory");
            asm volatile("s_wait_asynccnt 0" ::: "memory");
        }
        if (s + 1 < S_WORDS)
            __builtin_prefetch(&Gx[(size_t)(s + 1) * 2048 + g * 8], 0, 3);
        __syncthreads();
        {   // 32 dot products of length 512: 8 lanes per column
            int lc = tid >> 3, q = tid & 7;
            const __bf16* wr = &Wl[lc][q * 64];
            const __bf16* hr = &hbuf[q * 64];
            float sum = 0.0f;
#pragma unroll 8
            for (int k = 0; k < 64; ++k) sum += (float)wr[k] * (float)hr[k];
            partial[lc][q] = sum;
        }
        __syncthreads();
        if (tid < 32) {
            int lc = tid;
            int gc = (lc >> 3) * HW + g * 8 + (lc & 7);
            float d = partial[lc][0] + partial[lc][1] + partial[lc][2] + partial[lc][3] +
                      partial[lc][4] + partial[lc][5] + partial[lc][6] + partial[lc][7];
            gv[lc] = d + Gx[(size_t)s * 2048 + gc];
        }
        __syncthreads();
        if (tid < 8) {
            int j = g * 8 + tid;
            float i_ = sigmoidf(gv[tid]);
            float f_ = sigmoidf(gv[8 + tid]);
            float g_ = tanhf(gv[16 + tid]);
            float o_ = sigmoidf(gv[24 + tid]);
            float c  = f_ * cst[tid] + i_ * g_;
            float h  = o_ * tanhf(c);
            cst[tid] = c;
            hs_bf[(size_t)s * HW + j] = (__bf16)h;
        }
        __syncthreads();
        if (tid == 0) {                   // grid-wide step barrier
            __threadfence();
            atomicAdd(ctr, 1u);
            unsigned target = (unsigned)(NWG * (s + 1));
            while (atomicAdd(ctr, 0u) < target) __builtin_amdgcn_s_sleep(1);
            __threadfence();
        }
        __syncthreads();
    }
}

// ---------------------------------------------------------------------------
// Kernel 4: out = hs[1024x512] @ W_out^T + b_out   (WMMA, N padded 50->64)
// ---------------------------------------------------------------------------
__global__ __launch_bounds__(256) void out_kernel(
    const __bf16* __restrict__ hs_bf, const __bf16* __restrict__ Bo,
    const float* __restrict__ b_out, float* __restrict__ out) {
    int wid = blockIdx.x * 8 + (threadIdx.x >> 5);
    int lane = threadIdx.x & 31;
    int mt = wid >> 2, nt = wid & 3;      // 64 x 4 tiles
    v8f acc = {};
#pragma unroll
    for (int ks = 0; ks < 16; ++ks) {     // K = 512
        v16bf a = loadA_rm(hs_bf + (size_t)(mt * 16) * HW, HW, ks * 32);
        v16bf b = loadB_rm(Bo, 64, ks * 32, nt * 16);
        acc = __builtin_amdgcn_wmma_f32_16x16x32_bf16(
            false, a, false, b, (short)0, acc, false, false);
    }
    int n = lane & 15, half = lane >> 4;
#pragma unroll
    for (int r = 0; r < 8; ++r) {
        int m = mt * 16 + r + 8 * half, col = nt * 16 + n;
        if (col < TGT) out[(size_t)m * TGT + col] = acc[r] + b_out[col];
    }
}

// ---------------------------------------------------------------------------
extern "C" void kernel_launch(void* const* d_in, const int* in_sizes, int n_in,
                              void* d_out, int out_size, void* d_ws, size_t ws_size,
                              hipStream_t stream) {
    const int*   char_ids  = (const int*)d_in[0];
    const int*   char_lens = (const int*)d_in[1];
    const int*   word_ids  = (const int*)d_in[2];
    const float* char_emb  = (const float*)d_in[3];
    const float* word_emb  = (const float*)d_in[4];
    const float* Wc_ih     = (const float*)d_in[5];
    const float* Wc_hh     = (const float*)d_in[6];
    const float* bc        = (const float*)d_in[7];
    const float* Ww_ih     = (const float*)d_in[8];
    const float* Ww_hh     = (const float*)d_in[9];
    const float* bw        = (const float*)d_in[10];
    const float* W_out     = (const float*)d_in[11];
    const float* b_out     = (const float*)d_in[12];
    float* out = (float*)d_out;

    char* ws = (char*)d_ws;
    __bf16*   Bc     = (__bf16*)(ws + OFF_BC);
    __bf16*   Bwx    = (__bf16*)(ws + OFF_BWX);
    __bf16*   Bo     = (__bf16*)(ws + OFF_BO);
    __bf16*   WIN    = (__bf16*)(ws + OFF_WIN);
    float*    Gx     = (float*)(ws + OFF_GX);
    __bf16*   hs_bf  = (__bf16*)(ws + OFF_HS);
    unsigned* ctr    = (unsigned*)(ws + OFF_CTR);
    __bf16*   Whh_bf = (__bf16*)(ws + OFF_WHH);

    // weight prep + gathers (independent)
    prep_bc <<<(128 * 256 + 255) / 256, 256, 0, stream>>>(Wc_ih, Wc_hh, Bc);
    prep_bwx<<<(192 * 2048 + 255) / 256, 256, 0, stream>>>(Ww_ih, Bwx);
    prep_bo <<<(512 * 64 + 255) / 256, 256, 0, stream>>>(W_out, Bo);
    prep_win<<<(S_WORDS * DW + 255) / 256, 256, 0, stream>>>(word_ids, word_emb, WIN);
    prep_whh<<<(2048 * HW + 255) / 256, 256, 0, stream>>>(Ww_hh, Whh_bf);
    prep_ctr<<<1, 64, 0, stream>>>(ctr);

    // 1) char LSTM (64 independent 16-word tiles)
    char_lstm_kernel<<<S_WORDS / 16, 256, 0, stream>>>(char_ids, char_lens, char_emb,
                                                       Bc, bc, WIN);
    // 2) x-part gate precompute for word LSTM
    gx_kernel<<<(64 * 128) / 8, 256, 0, stream>>>(WIN, Bwx, bw, Gx);
    // 3) sequential word-LSTM recurrence (persistent cooperative grid, TDM+async)
    word_lstm_kernel<<<NWG, 256, 0, stream>>>(Whh_bf, Gx, hs_bf, ctr);
    // 4) output projection
    out_kernel<<<(64 * 4) / 8, 256, 0, stream>>>(hs_bf, Bo, b_out, out);
}
